// OnnxMultiheadRelativeAttention_65403761983790
// MI455X (gfx1250) — matrix operand order repaired and checked by reading, not compiled, not run on hardware
//
#include <hip/hip_runtime.h>

#define BATCH 4
#define TQ    2048
#define DM    1024
#define NH    16
#define DH    64

typedef __attribute__((ext_vector_type(16))) __bf16 v16bf;
typedef __attribute__((ext_vector_type(8)))  __bf16 v8bf;
typedef __attribute__((ext_vector_type(8)))  float  v8f;

union V16 { v16bf v; v8bf h[2]; };

__device__ __forceinline__ v8bf cvt8(float4 a, float4 b) {
    v8bf r;
    r[0] = (__bf16)a.x; r[1] = (__bf16)a.y; r[2] = (__bf16)a.z; r[3] = (__bf16)a.w;
    r[4] = (__bf16)b.x; r[5] = (__bf16)b.y; r[6] = (__bf16)b.z; r[7] = (__bf16)b.w;
    return r;
}

__device__ __forceinline__ v8f wmma_bf16(v16bf a, v16bf b, v8f c) {
    // D = A(16x32 bf16) * B(32x16 bf16) + C(16x16 f32)
    return __builtin_amdgcn_wmma_f32_16x16x32_bf16(false, a, false, b, (short)0, c,
                                                   false, false);
}

// ---------------------------------------------------------------------------
// Kernel 1: T5 relative position bias -> bias output region [H, Tq, Tk] f32
// ---------------------------------------------------------------------------
__global__ __launch_bounds__(256) void bias_kernel(const float* __restrict__ rel_emb,
                                                   float* __restrict__ bias_out) {
    unsigned int idx = blockIdx.x * 256u + threadIdx.x;   // < 16*2048*2048 = 2^26
    int kj = idx & (TQ - 1);
    int qi = (idx >> 11) & (TQ - 1);
    int h  = idx >> 22;
    int rp = kj - qi;                       // memory_pos - context_pos
    int buckets = (rp > 0) ? 16 : 0;        // NUM_BUCKETS/2 = 16
    int arp = (rp < 0) ? -rp : rp;
    int rl;
    if (arp < 8) {                          // max_exact = 8
        rl = arp;
    } else {
        // 8 + log(arp/8)/log(16) * 8, truncated, capped at 15
        float lf = __logf((float)arp * 0.125f) * 0.36067376f; // 1/ln(16)
        rl = 8 + (int)(lf * 8.0f);
        rl = rl > 15 ? 15 : rl;
    }
    bias_out[idx] = rel_emb[(buckets + rl) * NH + h];
}

// ---------------------------------------------------------------------------
// Kernel 2: projection GEMM  Y = (X @ W^T) * scale, X fp32 [8192,1024],
// W fp32 [1024,1024]; Y bf16.
//   vtrans == 0 : Y layout [B, H, T, DH]   (Q, K)
//   vtrans == 1 : Y layout [B, H, DH, T]   (V, pre-transposed for P*V WMMA)
// Block 256 thr = 8 waves; wave tile 16(M) x 64(N); WG tile 32 x 256.
// ---------------------------------------------------------------------------
__global__ __launch_bounds__(256) void proj_gemm_kernel(const float* __restrict__ X,
                                                        const float* __restrict__ W,
                                                        __bf16* __restrict__ Y,
                                                        float scale, int vtrans) {
    const int tid  = threadIdx.x;
    const int wave = tid >> 5, lane = tid & 31;
    const int l16  = lane & 15, half = lane >> 4;
    const int waveM = wave & 1, waveN = wave >> 1;
    const int mBase = blockIdx.x * 32 + waveM * 16;
    const int nBase = blockIdx.y * 256 + waveN * 64;

    v8f acc[4];
    #pragma unroll
    for (int t = 0; t < 4; ++t)
        #pragma unroll
        for (int r = 0; r < 8; ++r) acc[t][r] = 0.0f;

    const float* arow = X + (size_t)(mBase + l16) * DM;
    const int c0 = half * 16;
    for (int kb = 0; kb < DM; kb += 32) {
        __builtin_prefetch(arow + kb + 256, 0, 0);   // stream A ahead
        V16 a;
        {
            const float* p = arow + kb + c0;
            a.h[0] = cvt8(*(const float4*)(p),      *(const float4*)(p + 4));
            a.h[1] = cvt8(*(const float4*)(p + 8), *(const float4*)(p + 12));
        }
        #pragma unroll
        for (int nt = 0; nt < 4; ++nt) {
            const float* wrow = W + (size_t)(nBase + nt * 16 + l16) * DM + kb + half * 16;
            V16 bfr;
            bfr.h[0] = cvt8(*(const float4*)(wrow),     *(const float4*)(wrow + 4));
            bfr.h[1] = cvt8(*(const float4*)(wrow + 8), *(const float4*)(wrow + 12));
            acc[nt] = wmma_bf16(a.v, bfr.v, acc[nt]);
        }
    }

    #pragma unroll
    for (int r = 0; r < 8; ++r) {
        const int m = mBase + r + 8 * half;
        const int b = m >> 11, t = m & (TQ - 1);
        #pragma unroll
        for (int nt = 0; nt < 4; ++nt) {
            const int n  = nBase + nt * 16 + l16;
            const int h  = n >> 6, dh = n & 63;
            const __bf16 val = (__bf16)(acc[nt][r] * scale);
            if (vtrans)
                Y[(((size_t)(b * NH + h)) * DH + dh) * TQ + t] = val;
            else
                Y[(((size_t)(b * NH + h)) * TQ + t) * DH + dh] = val;
        }
    }
}

// ---------------------------------------------------------------------------
// Kernel 3: flash attention. Q pre-scaled by 1/8; bias added from d_out region.
// 128 thr = 4 waves; one wave per 16-row Q tile; online softmax over Tk.
// K layout [B,H,T,DH]; V layout [B,H,DH,T] (transposed -> vector B-frag loads).
// ---------------------------------------------------------------------------
__global__ __launch_bounds__(128) void attn_kernel(const __bf16* __restrict__ qws,
                                                   const __bf16* __restrict__ kws,
                                                   const __bf16* __restrict__ vws,
                                                   const float* __restrict__ bias,
                                                   __bf16* __restrict__ ctx) {
    const int tid  = threadIdx.x;
    const int wave = tid >> 5, lane = tid & 31;
    const int l16  = lane & 15, half = lane >> 4;

    const int qb    = blockIdx.x & 31;       // T/64 blocks
    const int bh    = blockIdx.x >> 5;       // b*16 + h
    const int h     = bh & 15;
    const int b     = bh >> 4;
    const int qBase = qb * 64 + wave * 16;

    const __bf16* qp = qws + (size_t)bh * TQ * DH;
    const __bf16* kp = kws + (size_t)bh * TQ * DH;
    const __bf16* vp = vws + (size_t)bh * DH * TQ;   // [DH][TQ]
    const float*  bp = bias + ((size_t)h * TQ + qBase) * TQ;

    __shared__ __bf16 pstage[4][16 * 32];
    __bf16* pb = pstage[wave];

    // Preload Q A-fragments for d = 0..31 and 32..63
    V16 aq[2];
    {
        const __bf16* qrow = qp + (size_t)(qBase + l16) * DH;
        const int c0 = half * 16;
        #pragma unroll
        for (int kt = 0; kt < 2; ++kt) {
            aq[kt].h[0] = *(const v8bf*)(qrow + kt * 32 + c0);
            aq[kt].h[1] = *(const v8bf*)(qrow + kt * 32 + c0 + 8);
        }
    }

    v8f o[4];
    float m_i[8], l_i[8];
    #pragma unroll
    for (int t = 0; t < 4; ++t)
        #pragma unroll
        for (int r = 0; r < 8; ++r) o[t][r] = 0.0f;
    #pragma unroll
    for (int r = 0; r < 8; ++r) { m_i[r] = -1e30f; l_i[r] = 0.0f; }

    for (int jb = 0; jb < TQ; jb += 32) {
        // prefetch next K block rows for this lane
        __builtin_prefetch(kp + (size_t)(jb + 32 + l16) * DH, 0, 0);

        // ---- scores: two 16x16 tiles (k columns jb..jb+15, jb+16..jb+31) ----
        v8f s[2];
        #pragma unroll
        for (int t = 0; t < 2; ++t) {
            #pragma unroll
            for (int r = 0; r < 8; ++r) s[t][r] = 0.0f;
            #pragma unroll
            for (int kt = 0; kt < 2; ++kt) {   // d = 0..31, 32..63
                V16 bk;
                const __bf16* krow =
                    kp + (size_t)(jb + t * 16 + l16) * DH + kt * 32 + half * 16;
                bk.h[0] = *(const v8bf*)(krow);
                bk.h[1] = *(const v8bf*)(krow + 8);
                s[t] = wmma_bf16(aq[kt].v, bk.v, s[t]);
            }
            #pragma unroll
            for (int r = 0; r < 8; ++r)
                s[t][r] += bp[(size_t)(r + 8 * half) * TQ + jb + t * 16 + l16];
        }

        // ---- online softmax (8 rows per lane, reduce across 16-lane half) ----
        float osc[8];
        #pragma unroll
        for (int r = 0; r < 8; ++r) {
            float mx = fmaxf(s[0][r], s[1][r]);
            #pragma unroll
            for (int msk = 1; msk < 16; msk <<= 1)
                mx = fmaxf(mx, __shfl_xor(mx, msk, 32));
            const float mnew = fmaxf(m_i[r], mx);
            osc[r] = __expf(m_i[r] - mnew);
            m_i[r] = mnew;
            const float p0 = __expf(s[0][r] - mnew);
            const float p1 = __expf(s[1][r] - mnew);
            s[0][r] = p0; s[1][r] = p1;
            float rs = p0 + p1;
            #pragma unroll
            for (int msk = 1; msk < 16; msk <<= 1)
                rs += __shfl_xor(rs, msk, 32);
            l_i[r] = l_i[r] * osc[r] + rs;
        }
        #pragma unroll
        for (int t = 0; t < 4; ++t)
            #pragma unroll
            for (int r = 0; r < 8; ++r) o[t][r] *= osc[r];

        // ---- re-layout P (C-layout -> A-layout) through LDS ----
        #pragma unroll
        for (int t = 0; t < 2; ++t)
            #pragma unroll
            for (int r = 0; r < 8; ++r)
                pb[(r + 8 * half) * 32 + t * 16 + l16] = (__bf16)s[t][r];
        asm volatile("s_wait_dscnt 0" ::: "memory");

        V16 ap;
        {
            const __bf16* prow = pb + l16 * 32;
            const int c0 = half * 16;
            ap.h[0] = *(const v8bf*)(prow + c0);
            ap.h[1] = *(const v8bf*)(prow + c0 + 8);
        }

        // ---- O += P * V : four 16x16 output tiles over d (V pre-transposed) ----
        #pragma unroll
        for (int t = 0; t < 4; ++t) {
            V16 bv;
            const __bf16* vrow = vp + (size_t)(t * 16 + l16) * TQ + jb + half * 16;
            bv.h[0] = *(const v8bf*)(vrow);
            bv.h[1] = *(const v8bf*)(vrow + 8);
            o[t] = wmma_bf16(ap.v, bv.v, o[t]);
        }
    }

    // ---- epilogue: normalize, write ctx bf16 in [B, T, H*DH] layout ----
    #pragma unroll
    for (int r = 0; r < 8; ++r) {
        const float inv = 1.0f / l_i[r];
        const int qrow = qBase + r + 8 * half;
        __bf16* crow = ctx + ((size_t)b * TQ + qrow) * DM + h * DH;
        #pragma unroll
        for (int t = 0; t < 4; ++t)
            crow[t * 16 + l16] = (__bf16)(o[t][r] * inv);
    }
}

// ---------------------------------------------------------------------------
// Kernel 4: output GEMM  out = ctx(bf16) @ Wo^T  -> fp32 [B*T, DM]
// ---------------------------------------------------------------------------
__global__ __launch_bounds__(256) void out_gemm_kernel(const __bf16* __restrict__ A,
                                                       const float* __restrict__ W,
                                                       float* __restrict__ Y) {
    const int tid  = threadIdx.x;
    const int wave = tid >> 5, lane = tid & 31;
    const int l16  = lane & 15, half = lane >> 4;
    const int waveM = wave & 1, waveN = wave >> 1;
    const int mBase = blockIdx.x * 32 + waveM * 16;
    const int nBase = blockIdx.y * 256 + waveN * 64;

    v8f acc[4];
    #pragma unroll
    for (int t = 0; t < 4; ++t)
        #pragma unroll
        for (int r = 0; r < 8; ++r) acc[t][r] = 0.0f;

    const __bf16* arow = A + (size_t)(mBase + l16) * DM;
    const int c0 = half * 16;
    for (int kb = 0; kb < DM; kb += 32) {
        __builtin_prefetch(arow + kb + 256, 0, 0);   // stream A ahead
        V16 a;
        a.h[0] = *(const v8bf*)(arow + kb + c0);
        a.h[1] = *(const v8bf*)(arow + kb + c0 + 8);
        #pragma unroll
        for (int nt = 0; nt < 4; ++nt) {
            const float* wrow = W + (size_t)(nBase + nt * 16 + l16) * DM + kb + half * 16;
            V16 bfr;
            bfr.h[0] = cvt8(*(const float4*)(wrow),     *(const float4*)(wrow + 4));
            bfr.h[1] = cvt8(*(const float4*)(wrow + 8), *(const float4*)(wrow + 12));
            acc[nt] = wmma_bf16(a.v, bfr.v, acc[nt]);
        }
    }

    #pragma unroll
    for (int r = 0; r < 8; ++r) {
        const int m = mBase + r + 8 * half;
        #pragma unroll
        for (int nt = 0; nt < 4; ++nt)
            Y[(size_t)m * DM + nBase + nt * 16 + l16] = acc[nt][r];
    }
}

// ---------------------------------------------------------------------------
extern "C" void kernel_launch(void* const* d_in, const int* in_sizes, int n_in,
                              void* d_out, int out_size, void* d_ws, size_t ws_size,
                              hipStream_t stream) {
    const float* q   = (const float*)d_in[0];
    const float* k   = (const float*)d_in[1];
    const float* v   = (const float*)d_in[2];
    const float* Wq  = (const float*)d_in[3];
    const float* Wk  = (const float*)d_in[4];
    const float* Wv  = (const float*)d_in[5];
    const float* Wo  = (const float*)d_in[6];
    const float* rel = (const float*)d_in[7];

    float* out  = (float*)d_out;
    float* bias = out + (size_t)BATCH * TQ * DM;        // second output tensor

    const size_t HSZ = (size_t)BATCH * TQ * DM;          // 8,388,608 bf16 elems
    __bf16* ws  = (__bf16*)d_ws;                         // 64 MB used
    __bf16* qws = ws;
    __bf16* kws = ws + HSZ;
    __bf16* vws = ws + 2 * HSZ;
    __bf16* ctx = ws + 3 * HSZ;

    // 1) position bias (also consumed by attention)
    bias_kernel<<<(NH * TQ * TQ) / 256, 256, 0, stream>>>(rel, bias);

    // 2) Q/K/V projections (Q pre-scaled by 1/sqrt(DH); V stored transposed)
    dim3 g(DM * 8 / 32, DM / 256);                       // (256, 4)
    proj_gemm_kernel<<<g, 256, 0, stream>>>(q, Wq, qws, 0.125f, 0);
    proj_gemm_kernel<<<g, 256, 0, stream>>>(k, Wk, kws, 1.0f, 0);
    proj_gemm_kernel<<<g, 256, 0, stream>>>(v, Wv, vws, 1.0f, 1);

    // 3) flash attention with bias
    attn_kernel<<<BATCH * NH * (TQ / 64), 128, 0, stream>>>(qws, kws, vws, bias, ctx);

    // 4) output projection
    out_gemm_kernel<<<g, 256, 0, stream>>>(ctx, Wo, out);
}
